// BigramNeuralNetwork_26895085207780
// MI455X (gfx1250) — matrix operand order repaired and checked by reading, not compiled, not run on hardware
//
#include <hip/hip_runtime.h>
#include <math.h>

// ---------------- model dims ----------------
namespace cfg {
constexpr int V = 32000, D = 512, DI = 1024, DS = 16, DC = 4, DTR = 32;
constexpr int L = 4, H = 2048, B = 2, T = 1024;
constexpr int M = B * T;                 // 2048 token rows
constexpr int XE = DTR + 2 * DS;         // 64, x_dbl row length
}

typedef __attribute__((ext_vector_type(16))) _Float16 v16h;
typedef __attribute__((ext_vector_type(8)))  _Float16 v8h;
typedef __attribute__((ext_vector_type(4)))  _Float16 v4h;
typedef __attribute__((ext_vector_type(8)))  float    v8f;
typedef __attribute__((ext_vector_type(4)))  float    v4f;
typedef __attribute__((ext_vector_type(4)))  int      v4i;

enum { EP_NONE = 0, EP_BIAS = 1, EP_BIAS_RELU = 2, EP_RES = 3, EP_RES_BIAS = 4,
       EP_SOFTPLUS_BIAS = 5 };

// ---------------- gfx1250 async global->LDS support (probe via __has_builtin) ----
#if defined(__gfx1250__) && __has_builtin(__builtin_amdgcn_global_load_async_to_lds_b128)
#define ASYNC_OK 1
#else
#define ASYNC_OK 0
#endif

#if ASYNC_OK
__device__ __forceinline__ void async_cp16(const _Float16* g, _Float16* l)
{
    // signature (from diagnostic): (global v4i* src, shared v4i* dst, imm off, imm cpol)
    __builtin_amdgcn_global_load_async_to_lds_b128(
        (__attribute__((address_space(1))) v4i*)g,
        (__attribute__((address_space(3))) v4i*)l, 0, 0);
}
template <int Ncnt>
__device__ __forceinline__ void s_wait_async()
{
#if __has_builtin(__builtin_amdgcn_s_wait_asynccnt)
    __builtin_amdgcn_s_wait_asynccnt(Ncnt);
#else
    asm volatile("s_wait_asynccnt %0" :: "n"(Ncnt) : "memory");
#endif
}
#endif

// =====================================================================
// WMMA GEMM on f16 operands: out[m,n] = epilogue( sum_k A16[m,k]*B16[n,k] )
// A16: (M x K) f16 row-major, lda;  B16: (N x K) f16 row-major (weights).
// Block 256 thr = 8 waves; tile 128(M) x 128(N) x 32(K); f32 accumulate.
// Async path: LDS double-buffer fed by global_load_async_to_lds_b128,
//             gated by s_wait_asynccnt (ASYNCcnt) + barriers.
// Fallback:  register double-buffer (load k+1 while WMMA on k).
// =====================================================================
template <int EP, bool BOUNDED>
__global__ __launch_bounds__(256)
void gemm_wmma(const _Float16* __restrict__ A16, int lda,
               const _Float16* __restrict__ B16, int ldb,
               const float* __restrict__ bias,
               const float* __restrict__ res,
               float* __restrict__ out, int ldc,
               int M, int N, int K)
{
    constexpr int BM = 128, BN = 128, BK = 32, LDT = BK + 8; // pad 8 halves
#if ASYNC_OK
    __shared__ _Float16 As[2][BM * LDT];
    __shared__ _Float16 Bs[2][BN * LDT];
#else
    __shared__ _Float16 As[1][BM * LDT];
    __shared__ _Float16 Bs[1][BN * LDT];
#endif

    const int tid  = threadIdx.x;
    const int lane = tid & 31;
    const int wid  = tid >> 5;      // 0..7
    const int wm   = wid & 3;       // 4 waves along M
    const int wn   = wid >> 2;      // 2 waves along N
    const int r16  = lane & 15;
    const int hsel = lane >> 4;     // 0/1 half-wave selector

    const int m0 = blockIdx.y * BM;
    const int n0 = blockIdx.x * BN;

    v8f acc[2][4];
    #pragma unroll
    for (int i = 0; i < 2; ++i)
        #pragma unroll
        for (int j = 0; j < 4; ++j)
            acc[i][j] = (v8f){0.f, 0.f, 0.f, 0.f, 0.f, 0.f, 0.f, 0.f};

    // staging: each thread moves 2x16B chunks per matrix per K-tile
    const int srow = tid >> 2;        // 0..63 (+64 on second pass)
    const int scol = (tid & 3) * 8;   // 0,8,16,24 halves

    auto a_gp = [&](int kt, int p) {
        return A16 + (size_t)(m0 + srow + 64 * p) * lda + kt * BK + scol;
    };
    auto b_gp = [&](int kt, int p) {
        int gn = n0 + srow + 64 * p;
        if (BOUNDED && gn >= N) gn = N - 1;   // clamp: cols >= N never stored
        return B16 + (size_t)gn * ldb + kt * BK + scol;
    };
    const int aoff0 = srow * LDT + scol;

    auto compute_tile = [&](const _Float16* Asb, const _Float16* Bsb) {
        v16h afrag[2];
        #pragma unroll
        for (int i = 0; i < 2; ++i) {
            const _Float16* ap = Asb + (wm * 32 + i * 16 + r16) * LDT;
            v8h a0 = *(const v8h*)(ap + hsel * 8);       // K 0-7 / 8-15
            v8h a1 = *(const v8h*)(ap + 16 + hsel * 8);  // K 16-23 / 24-31
            v16h a;
            #pragma unroll
            for (int e = 0; e < 8; ++e) { a[e] = a0[e]; a[8 + e] = a1[e]; }
            afrag[i] = a;
        }
        v16h bfrag[4];
        #pragma unroll
        for (int j = 0; j < 4; ++j) {
            const _Float16* bp = Bsb + (wn * 64 + j * 16 + r16) * LDT + hsel * 8;
            v8h b0 = *(const v8h*)(bp);                  // K 0-7 / 8-15
            v8h b1 = *(const v8h*)(bp + 16);
            v16h b;
            #pragma unroll
            for (int e = 0; e < 8; ++e) { b[e] = b0[e]; b[8 + e] = b1[e]; }
            bfrag[j] = b;
        }
        #pragma unroll
        for (int i = 0; i < 2; ++i)
            #pragma unroll
            for (int j = 0; j < 4; ++j)
                acc[i][j] = __builtin_amdgcn_wmma_f32_16x16x32_f16(
                    false, afrag[i], false, bfrag[j], (short)0, acc[i][j],
                    false, false);
    };

    const int ktiles = K / BK;

#if ASYNC_OK
    // ---- async double-buffered pipeline: 4 async issues (wave ops) per tile ----
    auto issue = [&](int kt, int buf) {
        #pragma unroll
        for (int p = 0; p < 2; ++p) {
            async_cp16(a_gp(kt, p), &As[buf][aoff0 + 64 * p * LDT]);
            async_cp16(b_gp(kt, p), &Bs[buf][aoff0 + 64 * p * LDT]);
        }
    };
    issue(0, 0);
    if (ktiles > 1) issue(1, 1);
    for (int kt = 0; kt < ktiles; ++kt) {
        if (kt < ktiles - 1) s_wait_async<4>();   // keep next tile in flight
        else                 s_wait_async<0>();
        __syncthreads();                          // tile kt fully in LDS
        compute_tile(As[kt & 1], Bs[kt & 1]);
        __syncthreads();                          // buffer free for reuse
        if (kt + 2 < ktiles) issue(kt + 2, kt & 1);
    }
#else
    // ---- register double-buffered pipeline ----
    v8h ra[2], rb[2];
    #pragma unroll
    for (int p = 0; p < 2; ++p) { ra[p] = *(const v8h*)a_gp(0, p);
                                  rb[p] = *(const v8h*)b_gp(0, p); }
    for (int kt = 0; kt < ktiles; ++kt) {
        #pragma unroll
        for (int p = 0; p < 2; ++p) {
            *(v8h*)(&As[0][aoff0 + 64 * p * LDT]) = ra[p];
            *(v8h*)(&Bs[0][aoff0 + 64 * p * LDT]) = rb[p];
        }
        __syncthreads();
        if (kt + 1 < ktiles) {          // overlap next-tile loads with WMMAs
            #pragma unroll
            for (int p = 0; p < 2; ++p) { ra[p] = *(const v8h*)a_gp(kt + 1, p);
                                          rb[p] = *(const v8h*)b_gp(kt + 1, p); }
        }
        if (kt + 2 < ktiles) {
            __builtin_prefetch((const void*)a_gp(kt + 2, 0), 0, 1);
            __builtin_prefetch((const void*)b_gp(kt + 2, 0), 0, 1);
        }
        compute_tile(As[0], Bs[0]);
        __syncthreads();
    }
#endif

    // Epilogue + store. C layout: VGPR r, lanes 0-15 -> M=r; lanes 16-31 -> M=8+r.
    #pragma unroll
    for (int i = 0; i < 2; ++i) {
        #pragma unroll
        for (int j = 0; j < 4; ++j) {
            const int n = n0 + wn * 64 + j * 16 + r16;
            if (BOUNDED && n >= N) continue;
            const int mb = m0 + wm * 32 + i * 16 + hsel * 8;
            float bv = 0.f;
            if (EP == EP_BIAS || EP == EP_BIAS_RELU || EP == EP_RES_BIAS ||
                EP == EP_SOFTPLUS_BIAS) bv = bias[n];
            #pragma unroll
            for (int r = 0; r < 8; ++r) {
                const size_t off = (size_t)(mb + r) * ldc + n;
                float v = acc[i][j][r];
                if (EP == EP_NONE)            out[off] = v;
                else if (EP == EP_BIAS)       out[off] = v + bv;
                else if (EP == EP_BIAS_RELU)  out[off] = fmaxf(v + bv, 0.f);
                else if (EP == EP_RES)        out[off] = res[off] + v;
                else if (EP == EP_RES_BIAS)   out[off] = res[off] + v + bv;
                else {                        // softplus(v + bias)
                    float x = v + bv;
                    out[off] = (x > 20.f) ? x : log1pf(__expf(x));
                }
            }
        }
    }
}

// ---------------- f32 -> f16 conversion (vectorized, n % 4 == 0) ----------------
__global__ __launch_bounds__(256)
void cvt16_kernel(const float* __restrict__ s, _Float16* __restrict__ d, int n4)
{
    int i = blockIdx.x * 256 + threadIdx.x;
    if (i >= n4) return;
    v4f f = ((const v4f*)s)[i];
    ((v4h*)d)[i] = (v4h){(_Float16)f.x, (_Float16)f.y, (_Float16)f.z, (_Float16)f.w};
}

// ---------------- embedding: x = tok_emb[idx] + pos_emb[:T] ----------------
__global__ __launch_bounds__(256)
void embed_kernel(const int* __restrict__ idx, const float* __restrict__ tok,
                  const float* __restrict__ pos, float* __restrict__ x)
{
    int i = blockIdx.x * 256 + threadIdx.x;
    if (i >= cfg::M * cfg::D) return;
    int d = i % cfg::D;
    int m = i / cfg::D;
    int t = m % cfg::T;
    x[i] = tok[(size_t)idx[m] * cfg::D + d] + pos[(size_t)t * cfg::D + d];
}

// ---------------- layernorm over D=512, one row per block ----------------
__global__ __launch_bounds__(256)
void layernorm_kernel(const float* __restrict__ x, const float* __restrict__ g,
                      const float* __restrict__ b, float* __restrict__ out)
{
    const int row = blockIdx.x;
    const int tid = threadIdx.x;
    const float* xr = x + (size_t)row * cfg::D;
    float v0 = xr[tid], v1 = xr[tid + 256];

    __shared__ float red1[8], red2[8];
    float s = v0 + v1;
    #pragma unroll
    for (int off = 16; off; off >>= 1) s += __shfl_xor(s, off, 32);
    if ((tid & 31) == 0) red1[tid >> 5] = s;
    __syncthreads();
    float tot = 0.f;
    #pragma unroll
    for (int i = 0; i < 8; ++i) tot += red1[i];
    const float mean = tot * (1.f / cfg::D);

    float d0 = v0 - mean, d1 = v1 - mean;
    float q = d0 * d0 + d1 * d1;
    #pragma unroll
    for (int off = 16; off; off >>= 1) q += __shfl_xor(q, off, 32);
    if ((tid & 31) == 0) red2[tid >> 5] = q;
    __syncthreads();
    float qt = 0.f;
    #pragma unroll
    for (int i = 0; i < 8; ++i) qt += red2[i];
    const float rstd = rsqrtf(qt * (1.f / cfg::D) + 1e-5f);

    float* orow = out + (size_t)row * cfg::D;
    orow[tid]       = d0 * rstd * g[tid]       + b[tid];
    orow[tid + 256] = d1 * rstd * g[tid + 256] + b[tid + 256];
}

// -------- depthwise causal conv (DC=4) over time on u = xz[:, :DI], + silu --------
__global__ __launch_bounds__(256)
void conv_silu_kernel(const float* __restrict__ xz, const float* __restrict__ cw,
                      const float* __restrict__ cb, float* __restrict__ us)
{
    int i = blockIdx.x * 256 + threadIdx.x;     // over M*DI
    if (i >= cfg::M * cfg::DI) return;
    const int c = i % cfg::DI;
    const int m = i / cfg::DI;
    const int t = m % cfg::T;
    float acc = cb[c];
    #pragma unroll
    for (int k = 0; k < cfg::DC; ++k) {
        int tt = t - (cfg::DC - 1) + k;
        if (tt >= 0)
            acc += cw[c * cfg::DC + k] *
                   xz[(size_t)(m - (t - tt)) * (2 * cfg::DI) + c];
    }
    us[i] = acc / (1.f + __expf(-acc));         // silu
}

// -------- selective scan: one lane per channel, 16-wide state in registers --------
__global__ __launch_bounds__(256)
void scan_kernel(const float* __restrict__ dt, const float* __restrict__ us,
                 const float* __restrict__ xdbl, const float* __restrict__ A_log,
                 float* __restrict__ ys)
{
    const int c = blockIdx.x * 256 + threadIdx.x;  // channel
    const int b = blockIdx.y;
    float a[cfg::DS], h[cfg::DS];
    #pragma unroll
    for (int s = 0; s < cfg::DS; ++s) {
        a[s] = -__expf(A_log[(size_t)c * cfg::DS + s]);
        h[s] = 0.f;
    }
    for (int t = 0; t < cfg::T; ++t) {
        const int m = b * cfg::T + t;
        const float dtc = dt[(size_t)m * cfg::DI + c];
        const float ut  = us[(size_t)m * cfg::DI + c];
        const float* bc = xdbl + (size_t)m * cfg::XE + cfg::DTR; // B_t, C_t (uniform)
        const float du = dtc * ut;
        float acc = 0.f;
        #pragma unroll
        for (int s = 0; s < cfg::DS; ++s) {
            float hn = h[s] * __expf(dtc * a[s]) + du * bc[s];
            h[s] = hn;
            acc += hn * bc[cfg::DS + s];
        }
        ys[(size_t)m * cfg::DI + c] = acc;
    }
}

// -------- gating: y = (ys + u*D_p) * silu(z),  z = xz[:, DI:] --------
__global__ __launch_bounds__(256)
void gate_kernel(const float* __restrict__ ys, const float* __restrict__ us,
                 const float* __restrict__ xz, const float* __restrict__ Dp,
                 float* __restrict__ yg)
{
    int i = blockIdx.x * 256 + threadIdx.x;   // over M*DI
    if (i >= cfg::M * cfg::DI) return;
    const int c = i % cfg::DI;
    const int m = i / cfg::DI;
    const float z = xz[(size_t)m * (2 * cfg::DI) + cfg::DI + c];
    const float y = ys[i] + us[i] * Dp[c];
    yg[i] = y * (z / (1.f + __expf(-z)));
}

// =====================================================================
extern "C" void kernel_launch(void* const* d_in, const int* in_sizes, int n_in,
                              void* d_out, int out_size, void* d_ws, size_t ws_size,
                              hipStream_t stream)
{
    (void)in_sizes; (void)n_in; (void)out_size; (void)ws_size;
    using namespace cfg;

    const int*   idx    = (const int*)  d_in[0];
    const float* tok    = (const float*)d_in[1];
    const float* pos    = (const float*)d_in[2];
    const float* ln1_g  = (const float*)d_in[3];
    const float* ln1_b  = (const float*)d_in[4];
    const float* W_in   = (const float*)d_in[5];
    const float* conv_w = (const float*)d_in[6];
    const float* conv_b = (const float*)d_in[7];
    const float* W_xp   = (const float*)d_in[8];
    const float* W_dt   = (const float*)d_in[9];
    const float* b_dt   = (const float*)d_in[10];
    const float* A_log  = (const float*)d_in[11];
    const float* D_p    = (const float*)d_in[12];
    const float* W_out  = (const float*)d_in[13];
    const float* ln2_g  = (const float*)d_in[14];
    const float* ln2_b  = (const float*)d_in[15];
    const float* W_f1   = (const float*)d_in[16];
    const float* b_f1   = (const float*)d_in[17];
    const float* W_f2   = (const float*)d_in[18];
    const float* b_f2   = (const float*)d_in[19];
    const float* lm_w   = (const float*)d_in[20];
    const float* lm_b   = (const float*)d_in[21];
    float* logits = (float*)d_out;

    // workspace layout: f32 buffers, then f16 staging buffers
    float* ws   = (float*)d_ws;
    float* x    = ws;              ws += (size_t)M * D;
    float* ln   = ws;              ws += (size_t)M * D;
    float* xz   = ws;              ws += (size_t)M * 2 * DI;
    float* usil = ws;              ws += (size_t)M * DI;
    float* xdbl = ws;              ws += (size_t)M * XE;
    float* dtb  = ws;              ws += (size_t)M * DI;
    float* yscn = ws;              ws += (size_t)M * DI;
    float* ygat = ws;              ws += (size_t)M * DI;
    float* hffn = ws;              ws += (size_t)M * H;

    _Float16* hb    = (_Float16*)ws;
    _Float16* a16   = hb;          hb += (size_t)M * H;        // largest A (M x 2048)
    _Float16* wb16  = hb;          hb += (size_t)2 * DI * D;   // largest layer weight
    _Float16* lmw16 = hb;          hb += (size_t)V * D;

    const dim3 blk(256);
    auto ggrid = [](int N_, int M_) { return dim3((N_ + 127) / 128, M_ / 128); };
    auto cvt = [&](const float* s, _Float16* d, size_t n) {
        cvt16_kernel<<<dim3((unsigned)((n / 4 + 255) / 256)), blk, 0, stream>>>(s, d, (int)(n / 4));
    };

    // embedding
    embed_kernel<<<dim3((M * D + 255) / 256), blk, 0, stream>>>(idx, tok, pos, x);

    for (int l = 0; l < L; ++l) {
        const float* Wi  = W_in  + (size_t)l * 2 * DI * D;
        const float* cw  = conv_w + (size_t)l * DI * DC;
        const float* cb  = conv_b + (size_t)l * DI;
        const float* Wxp = W_xp  + (size_t)l * XE * DI;
        const float* Wdt = W_dt  + (size_t)l * DI * DTR;
        const float* bdt = b_dt  + (size_t)l * DI;
        const float* Al  = A_log + (size_t)l * DI * DS;
        const float* Dpl = D_p   + (size_t)l * DI;
        const float* Wo  = W_out + (size_t)l * D * DI;
        const float* W1  = W_f1  + (size_t)l * H * D;
        const float* bf1 = b_f1  + (size_t)l * H;
        const float* W2  = W_f2  + (size_t)l * D * H;
        const float* bf2 = b_f2  + (size_t)l * D;

        // ln1 ; xz = ln @ W_in^T
        layernorm_kernel<<<dim3(M), blk, 0, stream>>>(x, ln1_g + l * D, ln1_b + l * D, ln);
        cvt(ln, a16, (size_t)M * D);
        cvt(Wi, wb16, (size_t)2 * DI * D);
        gemm_wmma<EP_NONE, false><<<ggrid(2 * DI, M), blk, 0, stream>>>(
            a16, D, wb16, D, nullptr, nullptr, xz, 2 * DI, M, 2 * DI, D);
        // depthwise conv + silu -> u
        conv_silu_kernel<<<dim3((M * DI + 255) / 256), blk, 0, stream>>>(xz, cw, cb, usil);
        // x_dbl = u @ W_xp^T   (N=64 -> BOUNDED)
        cvt(usil, a16, (size_t)M * DI);
        cvt(Wxp, wb16, (size_t)XE * DI);
        gemm_wmma<EP_NONE, true><<<ggrid(XE, M), blk, 0, stream>>>(
            a16, DI, wb16, DI, nullptr, nullptr, xdbl, XE, M, XE, DI);
        // dt = softplus(x_dbl[:, :32] @ W_dt^T + b_dt)
        cvt(xdbl, a16, (size_t)M * XE);
        cvt(Wdt, wb16, (size_t)DI * DTR);
        gemm_wmma<EP_SOFTPLUS_BIAS, false><<<ggrid(DI, M), blk, 0, stream>>>(
            a16, XE, wb16, DTR, bdt, nullptr, dtb, DI, M, DI, DTR);
        // selective scan + gating
        scan_kernel<<<dim3(DI / 256, B), blk, 0, stream>>>(dtb, usil, xdbl, Al, yscn);
        gate_kernel<<<dim3((M * DI + 255) / 256), blk, 0, stream>>>(yscn, usil, xz, Dpl, ygat);
        // x += y @ W_out^T
        cvt(ygat, a16, (size_t)M * DI);
        cvt(Wo, wb16, (size_t)D * DI);
        gemm_wmma<EP_RES, false><<<ggrid(D, M), blk, 0, stream>>>(
            a16, DI, wb16, DI, nullptr, x, x, D, M, D, DI);
        // ln2 + FFN
        layernorm_kernel<<<dim3(M), blk, 0, stream>>>(x, ln2_g + l * D, ln2_b + l * D, ln);
        cvt(ln, a16, (size_t)M * D);
        cvt(W1, wb16, (size_t)H * D);
        gemm_wmma<EP_BIAS_RELU, false><<<ggrid(H, M), blk, 0, stream>>>(
            a16, D, wb16, D, bf1, nullptr, hffn, H, M, H, D);
        cvt(hffn, a16, (size_t)M * H);
        cvt(W2, wb16, (size_t)D * H);
        gemm_wmma<EP_RES_BIAS, false><<<ggrid(D, M), blk, 0, stream>>>(
            a16, H, wb16, H, bf2, x, x, D, M, D, H);
    }

    // logits = x @ lm_w^T + lm_b   (dominant GEMM: 2048 x 32000 x 512)
    cvt(x, a16, (size_t)M * D);
    cvt(lm_w, lmw16, (size_t)V * D);
    gemm_wmma<EP_BIAS, false><<<ggrid(V, M), blk, 0, stream>>>(
        a16, D, lmw16, D, lm_b, nullptr, logits, V, M, V, D);
}
